// MultiModalMolecularModel_81114752352572
// MI455X (gfx1250) — matrix operand order repaired and checked by reading, not compiled, not run on hardware
//
#include <hip/hip_runtime.h>
#include <hip/hip_bf16.h>
#include <math.h>

// ---------------------------------------------------------------------------
// Types for CDNA5 WMMA (gfx1250, wave32)
// ---------------------------------------------------------------------------
typedef __attribute__((ext_vector_type(16))) __bf16 v16bf;
typedef __attribute__((ext_vector_type(8)))  __bf16 v8bf;
typedef __attribute__((ext_vector_type(8)))  float  v8f;

static __device__ __forceinline__ v16bf join16(v8bf lo, v8bf hi) {
  return __builtin_shufflevector(lo, hi, 0,1,2,3,4,5,6,7,8,9,10,11,12,13,14,15);
}

static __device__ __forceinline__ v8f wmma_bf16(v16bf a, v16bf b, v8f c) {
  // D(16x16,f32) = A(16x32,bf16) * B(32x16,bf16) + C
  return __builtin_amdgcn_wmma_f32_16x16x32_bf16(
      /*neg_a=*/false, a, /*neg_b=*/false, b,
      /*c_mod=*/(short)0, c, /*reuse_a=*/false, /*reuse_b=*/false);
}

#define HDIM 256
#define ATOM_TYPES 100
#define ATOM_PAD   112
#define BOND_TYPES 5
#define BOND_PAD   16

// ---------------------------------------------------------------------------
// Prep kernels
// ---------------------------------------------------------------------------

// Sinusoidal timestep embedding, [B][H] f32. grid=B, block=H.
__global__ void k_temb(const float* __restrict__ t, float* __restrict__ temb, int H) {
  int b = blockIdx.x;
  int j = threadIdx.x;
  int half = H >> 1;
  float tb = t[b];
  float lf = -logf(10000.0f) / (float)(half - 1);
  float v;
  if (j < half) v = sinf(tb * expf((float)j * lf));
  else          v = cosf(tb * expf((float)(j - half) * lf));
  temb[b * H + j] = v;
}

// h = node_features + temb[batch]  -> bf16 (for WMMA), also pass node_features to out.
__global__ void k_make_h(const float* __restrict__ nf, const int* __restrict__ batch,
                         const float* __restrict__ temb, __bf16* __restrict__ hbf,
                         float* __restrict__ nf_out, int total, int H) {
  int idx = blockIdx.x * blockDim.x + threadIdx.x;
  if (idx >= total) return;
  int n = idx / H;
  int j = idx - n * H;
  float x = nf[idx];
  float v = x + temb[batch[n] * H + j];
  hbf[idx] = (__bf16)v;
  nf_out[idx] = x;
}

__global__ void k_copy_f32(const float* __restrict__ src, float* __restrict__ dst, int n) {
  int idx = blockIdx.x * blockDim.x + threadIdx.x;
  if (idx < n) dst[idx] = src[idx];
}

// Transpose [K][Ncols] f32 weight -> [Npad][K] bf16 (zero pad cols >= Ncols).
__global__ void k_transpose_pad(const float* __restrict__ in, __bf16* __restrict__ out,
                                int K, int Ncols, int Npad) {
  int idx = blockIdx.x * blockDim.x + threadIdx.x;
  if (idx >= K * Npad) return;
  int n = idx / K;
  int k = idx - n * K;
  float v = (n < Ncols) ? in[k * Ncols + n] : 0.0f;
  out[n * K + k] = (__bf16)v;
}

// ---------------------------------------------------------------------------
// Atom MLP: logits = relu(h @ W1 + b1) @ W2 + b2
// 1 block = 16 nodes, 256 threads = 8 waves.
// ---------------------------------------------------------------------------
__global__ __launch_bounds__(256) void k_atom(
    const __bf16* __restrict__ hbf,   // [N][256] bf16
    const __bf16* __restrict__ w1T,   // [256][256] bf16 (row = out channel)
    const float*  __restrict__ b1,    // [256]
    const __bf16* __restrict__ w2T,   // [112][256] bf16 (zero padded)
    const float*  __restrict__ b2,    // [100]
    float* __restrict__ out,          // [N][100]
    int N) {
  __shared__ __align__(16) __bf16 sX[16 * HDIM];
  __shared__ __align__(16) __bf16 sHid[16 * HDIM];

  const int tid  = threadIdx.x;
  const int n0   = blockIdx.x * 16;
  const int wave = tid >> 5;
  const int lane = tid & 31;
  const int ln   = lane & 15;   // fragment row/col within tile
  const int kh   = lane >> 4;   // K-half selector

  // Stage 16 activation rows (each thread copies 16 bf16 = 2 x B128 load).
  {
    int r = tid >> 4;
    int c = tid & 15;
    int row = n0 + r; if (row >= N) row = N - 1;
    const __bf16* src = hbf + (size_t)row * HDIM + c * 16;
    *(v8bf*)&sX[r * HDIM + c * 16]     = *(const v8bf*)src;
    *(v8bf*)&sX[r * HDIM + c * 16 + 8] = *(const v8bf*)(src + 8);
  }
  __syncthreads();

  // ---- Layer 1: wave w computes hidden cols [32w, 32w+32) over K=256 ----
  v8f acc0 = {}; v8f acc1 = {};
  const int c0 = wave * 32 + ln;
  const int c1 = c0 + 16;
#pragma unroll
  for (int kb = 0; kb < HDIM; kb += 32) {
    v16bf a = join16(*(const v8bf*)&sX[ln * HDIM + kb + kh * 8],
                     *(const v8bf*)&sX[ln * HDIM + kb + 16 + kh * 8]);
    const __bf16* bp0 = w1T + (size_t)c0 * HDIM + kb + kh * 8;
    const __bf16* bp1 = w1T + (size_t)c1 * HDIM + kb + kh * 8;
    v16bf b0 = join16(*(const v8bf*)bp0, *(const v8bf*)(bp0 + 16));
    v16bf b1v = join16(*(const v8bf*)bp1, *(const v8bf*)(bp1 + 16));
    acc0 = wmma_bf16(a, b0, acc0);
    acc1 = wmma_bf16(a, b1v, acc1);
  }
  // bias + ReLU -> LDS (bf16)
  {
    float bb0 = b1[c0], bb1 = b1[c1];
#pragma unroll
    for (int j = 0; j < 8; ++j) {
      int r = j + kh * 8;
      float v0 = acc0[j] + bb0; v0 = v0 > 0.0f ? v0 : 0.0f;
      float v1 = acc1[j] + bb1; v1 = v1 > 0.0f ? v1 : 0.0f;
      sHid[r * HDIM + c0] = (__bf16)v0;
      sHid[r * HDIM + c1] = (__bf16)v1;
    }
  }
  __syncthreads();

  // ---- Layer 2: waves 0..6 compute 16-wide output tiles of the 112 padded cols ----
  if (wave < 7) {
    v8f acc = {};
    const int oc = wave * 16 + ln;
#pragma unroll
    for (int kb = 0; kb < HDIM; kb += 32) {
      v16bf a = join16(*(const v8bf*)&sHid[ln * HDIM + kb + kh * 8],
                       *(const v8bf*)&sHid[ln * HDIM + kb + 16 + kh * 8]);
      const __bf16* bp = w2T + (size_t)oc * HDIM + kb + kh * 8;
      v16bf b = join16(*(const v8bf*)bp, *(const v8bf*)(bp + 16));
      acc = wmma_bf16(a, b, acc);
    }
    float bb = (oc < ATOM_TYPES) ? b2[oc] : 0.0f;
#pragma unroll
    for (int j = 0; j < 8; ++j) {
      int r = j + kh * 8;
      if (oc < ATOM_TYPES && (n0 + r) < N)
        out[(size_t)(n0 + r) * ATOM_TYPES + oc] = acc[j] + bb;
    }
  }
}

// ---------------------------------------------------------------------------
// Bond MLP: logits = relu([h[row],h[col]] @ W1 + b1) @ W2 + b2
// 1 block = 64 edges (4 M-tiles), 256 threads = 8 waves.
// Each wave: cols [32w,32w+32), all 4 M-tiles -> 8 accumulators; each weight
// B-fragment is reused across 4 WMMAs (4x less L2 weight traffic).
// Dynamic LDS: 64KB gathered edge features + 32KB hidden = 96KB.
// ---------------------------------------------------------------------------
#define BOND_MT 4                      // M-tiles (of 16 edges) per block
#define BOND_EPB (BOND_MT * 16)        // 64 edges per block

__global__ __launch_bounds__(256) void k_bond(
    const __bf16* __restrict__ hbf,   // [N][256] bf16
    const int*    __restrict__ ei,    // edge_index [2][E]
    const __bf16* __restrict__ w1T,   // [256][512] bf16
    const float*  __restrict__ b1,    // [256]
    const __bf16* __restrict__ w2T,   // [16][256] bf16 (zero padded)
    const float*  __restrict__ b2,    // [5]
    float* __restrict__ out,          // [E][5]
    int E) {
  extern __shared__ __align__(16) char smem[];
  __bf16* sEF  = (__bf16*)smem;                          // [64][512] = 64 KB
  __bf16* sHid = (__bf16*)(smem + BOND_EPB * 512 * 2);   // [64][256] = 32 KB

  const int tid  = threadIdx.x;
  const int e0   = blockIdx.x * BOND_EPB;
  const int wave = tid >> 5;
  const int lane = tid & 31;
  const int ln   = lane & 15;
  const int kh   = lane >> 4;

  // Gather: 4 threads per edge, each copies 128 contiguous bf16 (16 x B128).
  {
    int e = tid >> 2;        // local edge 0..63
    int c = tid & 3;         // chunk 0..3 (2 from h[row], 2 from h[col])
    int edge = e0 + e; if (edge >= E) edge = E - 1;
    int node = (c < 2) ? ei[edge] : ei[E + edge];
    const __bf16* src = hbf + (size_t)node * HDIM + (c & 1) * 128;
    __bf16* dst = sEF + e * 512 + c * 128;
#pragma unroll
    for (int i = 0; i < 16; ++i)
      *(v8bf*)(dst + i * 8) = *(const v8bf*)(src + i * 8);
  }
  __syncthreads();

  // ---- Layer 1: wave w -> hidden cols [32w, 32w+32) for 4 M-tiles, K = 512 ----
  v8f acc[BOND_MT][2];
#pragma unroll
  for (int mt = 0; mt < BOND_MT; ++mt) { acc[mt][0] = (v8f){}; acc[mt][1] = (v8f){}; }
  const int c0 = wave * 32 + ln;
  const int c1 = c0 + 16;
#pragma unroll
  for (int kb = 0; kb < 512; kb += 32) {
    const __bf16* bp0 = w1T + (size_t)c0 * 512 + kb + kh * 8;
    const __bf16* bp1 = w1T + (size_t)c1 * 512 + kb + kh * 8;
    v16bf b0  = join16(*(const v8bf*)bp0, *(const v8bf*)(bp0 + 16));
    v16bf b1v = join16(*(const v8bf*)bp1, *(const v8bf*)(bp1 + 16));
#pragma unroll
    for (int mt = 0; mt < BOND_MT; ++mt) {
      const __bf16* ap = sEF + (mt * 16 + ln) * 512 + kb + kh * 8;
      v16bf a = join16(*(const v8bf*)ap, *(const v8bf*)(ap + 16));
      acc[mt][0] = wmma_bf16(a, b0,  acc[mt][0]);
      acc[mt][1] = wmma_bf16(a, b1v, acc[mt][1]);
    }
  }
  // bias + ReLU -> LDS (bf16)
  {
    float bb0 = b1[c0], bb1 = b1[c1];
#pragma unroll
    for (int mt = 0; mt < BOND_MT; ++mt) {
#pragma unroll
      for (int j = 0; j < 8; ++j) {
        int r = mt * 16 + j + kh * 8;
        float v0 = acc[mt][0][j] + bb0; v0 = v0 > 0.0f ? v0 : 0.0f;
        float v1 = acc[mt][1][j] + bb1; v1 = v1 > 0.0f ? v1 : 0.0f;
        sHid[r * HDIM + c0] = (__bf16)v0;
        sHid[r * HDIM + c1] = (__bf16)v1;
      }
    }
  }
  __syncthreads();

  // ---- Layer 2: waves 0..3 each handle one M-tile of 16 edges, K = 256 ----
  if (wave < BOND_MT) {
    v8f a2 = {};
#pragma unroll
    for (int kb = 0; kb < HDIM; kb += 32) {
      const __bf16* ap = sHid + (wave * 16 + ln) * HDIM + kb + kh * 8;
      v16bf a = join16(*(const v8bf*)ap, *(const v8bf*)(ap + 16));
      const __bf16* bp = w2T + (size_t)ln * HDIM + kb + kh * 8;
      v16bf b = join16(*(const v8bf*)bp, *(const v8bf*)(bp + 16));
      a2 = wmma_bf16(a, b, a2);
    }
    float bb = (ln < BOND_TYPES) ? b2[ln] : 0.0f;
#pragma unroll
    for (int j = 0; j < 8; ++j) {
      int r = wave * 16 + j + kh * 8;
      if (ln < BOND_TYPES && (e0 + r) < E)
        out[(size_t)(e0 + r) * BOND_TYPES + ln] = a2[j] + bb;
    }
  }
}

// ---------------------------------------------------------------------------
// Host launcher
// ---------------------------------------------------------------------------
extern "C" void kernel_launch(void* const* d_in, const int* in_sizes, int n_in,
                              void* d_out, int out_size, void* d_ws, size_t ws_size,
                              hipStream_t stream) {
  const float* node_features = (const float*)d_in[0];
  const float* pos_pred      = (const float*)d_in[1];
  const int*   edge_index    = (const int*)  d_in[2];
  const int*   batch         = (const int*)  d_in[3];
  const float* t_discrete    = (const float*)d_in[4];
  const float* atom_w1       = (const float*)d_in[5];
  const float* atom_b1       = (const float*)d_in[6];
  const float* atom_w2       = (const float*)d_in[7];
  const float* atom_b2       = (const float*)d_in[8];
  const float* bond_w1       = (const float*)d_in[9];
  const float* bond_b1       = (const float*)d_in[10];
  const float* bond_w2       = (const float*)d_in[11];
  const float* bond_b2       = (const float*)d_in[12];

  const int H = HDIM;
  const int N = in_sizes[0] / H;
  const int E = in_sizes[2] / 2;
  const int B = in_sizes[4];

  // Workspace carve-up (bytes, 256B aligned).
  char* ws = (char*)d_ws;
  size_t cur = 0;
  auto carve = [&](size_t bytes) { void* p = ws + cur; cur += (bytes + 255) & ~(size_t)255; return p; };
  float*  temb   = (float*) carve((size_t)B * H * 4);
  __bf16* hbf    = (__bf16*)carve((size_t)N * H * 2);
  __bf16* aw1T   = (__bf16*)carve((size_t)H * H * 2);
  __bf16* aw2T   = (__bf16*)carve((size_t)ATOM_PAD * H * 2);
  __bf16* bw1T   = (__bf16*)carve((size_t)H * 2 * H * 2);
  __bf16* bw2T   = (__bf16*)carve((size_t)BOND_PAD * H * 2);
  (void)ws_size;

  // Output layout: pos_pred | atom_logits | bond_logits | node_features
  float* out_pos  = (float*)d_out;
  float* out_atom = out_pos  + (size_t)N * 3;
  float* out_bond = out_atom + (size_t)N * ATOM_TYPES;
  float* out_nf   = out_bond + (size_t)E * BOND_TYPES;

  // --- prep ---
  k_temb<<<B, H, 0, stream>>>(t_discrete, temb, H);
  {
    int total = N * H;
    k_make_h<<<(total + 255) / 256, 256, 0, stream>>>(node_features, batch, temb,
                                                      hbf, out_nf, total, H);
  }
  {
    int total = N * 3;
    k_copy_f32<<<(total + 255) / 256, 256, 0, stream>>>(pos_pred, out_pos, total);
  }
  k_transpose_pad<<<(H * H + 255) / 256, 256, 0, stream>>>(atom_w1, aw1T, H, H, H);
  k_transpose_pad<<<(H * ATOM_PAD + 255) / 256, 256, 0, stream>>>(atom_w2, aw2T, H, ATOM_TYPES, ATOM_PAD);
  k_transpose_pad<<<(2 * H * H + 255) / 256, 256, 0, stream>>>(bond_w1, bw1T, 2 * H, H, H);
  k_transpose_pad<<<(H * BOND_PAD + 255) / 256, 256, 0, stream>>>(bond_w2, bw2T, H, BOND_TYPES, BOND_PAD);

  // --- WMMA MLPs ---
  k_atom<<<(N + 15) / 16, 256, 0, stream>>>(hbf, aw1T, atom_b1, aw2T, atom_b2, out_atom, N);
  {
    int blocks = (E + BOND_EPB - 1) / BOND_EPB;
    size_t shmem = (size_t)(BOND_EPB * 512 + BOND_EPB * HDIM) * 2;  // 96 KB
    k_bond<<<blocks, 256, shmem, stream>>>(hbf, edge_index, bw1T, bond_b1, bw2T, bond_b2, out_bond, E);
  }
}
